// MultiAttn_26568667693432
// MI455X (gfx1250) — matrix operand order repaired and checked by reading, not compiled, not run on hardware
//
#include <hip/hip_runtime.h>
#include <hip/hip_bf16.h>
#include <math.h>

typedef __attribute__((ext_vector_type(16))) _Float16 v16h;
typedef __attribute__((ext_vector_type(8)))  _Float16 v8h;
typedef __attribute__((ext_vector_type(8)))  float    v8f;

#define S_LEN 4096
#define D_MODEL 512
#define N_HEADS 8
#define D_HEAD 64
#define SROW 68   // padded LDS row stride (floats): 272B, 16B aligned, bank-safe
// Q scale: 1/sqrt(64) * log2(e)  -> softmax runs in exp2 domain (v_exp_f32 native)
#define Q_SCALE 0.1803368801111204f

static __device__ __forceinline__ v16h cat8(v8h lo, v8h hi) {
    return __builtin_shufflevector(lo, hi, 0,1,2,3,4,5,6,7,8,9,10,11,12,13,14,15);
}

static __device__ __forceinline__ v8f wmma_f16(v16h a, v16h b, v8f c) {
    return __builtin_amdgcn_wmma_f32_16x16x32_f16(false, a, false, b, (short)0, c, false, false);
}

// ---------------------------------------------------------------------------
// Kernel 0: streaming fp32 -> f16 conversion (8 elements / thread, b128 out)
// ---------------------------------------------------------------------------
__global__ __launch_bounds__(256)
void cvt_f32_f16_kernel(const float* __restrict__ src, _Float16* __restrict__ dst)
{
    const int base = (blockIdx.x * 256 + threadIdx.x) * 8;
    const float4 f0 = ((const float4*)(src + base))[0];
    const float4 f1 = ((const float4*)(src + base))[1];
    v8h o;
    o[0] = (_Float16)f0.x; o[1] = (_Float16)f0.y;
    o[2] = (_Float16)f0.z; o[3] = (_Float16)f0.w;
    o[4] = (_Float16)f1.x; o[5] = (_Float16)f1.y;
    o[6] = (_Float16)f1.z; o[7] = (_Float16)f1.w;
    *(v8h*)(dst + base) = o;
}

// ---------------------------------------------------------------------------
// Kernel 1: projection  Y = X @ W^T + b  (f16 operands, fp32 accum).
// One wave computes a 32x64 strip; all 12 loads per k-step batched before
// the 8 WMMAs (B fragments reused 2x, A fragments 4x).
// MODE: 0 = Q (scale Q_SCALE, [H][S][64]), 1 = K ([H][S][64]), 2 = V^T ([H][64][S])
// ---------------------------------------------------------------------------
template<int MODE>
__global__ __launch_bounds__(32)
void qkv_proj_kernel(const _Float16* __restrict__ X, const _Float16* __restrict__ W,
                     const float* __restrict__ bias, _Float16* __restrict__ out)
{
    const int lane = threadIdx.x & 31;
    const int hlf  = lane >> 4;
    const int l15  = lane & 15;
    const int m0   = blockIdx.x * 32;      // 0..127 (S/32)
    const int nb   = blockIdx.y;           // 0..7   (D/64)

    v8f acc0[4] = {}, acc1[4] = {};
    for (int k0 = 0; k0 < D_MODEL; k0 += 32) {
        const _Float16* ap = X + (m0 + l15) * D_MODEL + k0 + hlf * 16;
        const _Float16* ap1 = ap + 16 * D_MODEL;
        const _Float16* bp = W + (nb * 64 + l15) * D_MODEL + k0 + hlf * 16;
        const v16h a0 = cat8(*(const v8h*)ap,  *(const v8h*)(ap + 8));
        const v16h a1 = cat8(*(const v8h*)ap1, *(const v8h*)(ap1 + 8));
        v16h b[4];
#pragma unroll
        for (int t = 0; t < 4; ++t) b[t] = *(const v16h*)(bp + t * 16 * D_MODEL);
#pragma unroll
        for (int t = 0; t < 4; ++t) {
            acc0[t] = wmma_f16(a0, b[t], acc0[t]);
            acc1[t] = wmma_f16(a1, b[t], acc1[t]);
        }
    }

#pragma unroll
    for (int t = 0; t < 4; ++t) {
        const int   n  = nb * 64 + t * 16 + l15;   // C-layout col for this lane
        const float bb = bias[n];
        const int   h  = n >> 6;
        const int   dk = n & 63;
#pragma unroll
        for (int r = 0; r < 8; ++r) {
#pragma unroll
            for (int g = 0; g < 2; ++g) {
                const int row = m0 + g * 16 + r + 8 * hlf;
                float val = (g ? acc1[t][r] : acc0[t][r]) + bb;
                if (MODE == 0) val *= Q_SCALE;     // 1/sqrt(64) * log2(e)
                const _Float16 hv = (_Float16)val;
                if (MODE == 2) out[(h * D_HEAD + dk) * S_LEN + row] = hv;   // V^T
                else           out[h * S_LEN * D_HEAD + row * D_HEAD + dk] = hv;
            }
        }
    }
}

// ---------------------------------------------------------------------------
// Kernel 2: flash attention per head.  4 waves/block; each wave owns 16 query
// rows and streams 64-key tiles.  Next tile's K/V cachelines are prefetched
// (global_prefetch_b8: no LOADcnt, no registers) while softmax runs, so the
// score-phase waits hit the WGP cache.  Raw scores pass through LDS into
// A-fragment layout where a full row lives in 2 lanes: reduction = local
// VALU tree + ONE shfl_xor(16) each for max & sum.  Softmax runs in the
// exp2 domain (log2e folded into Q).
// Q,K: f16 [H][S][64]; VT: f16 [H][64][S]; O: f16 [S][512].
// ---------------------------------------------------------------------------
__global__ __launch_bounds__(128)
void flash_attn_kernel(const _Float16* __restrict__ Qh,
                       const _Float16* __restrict__ Kh,
                       const _Float16* __restrict__ VTh,
                       _Float16* __restrict__ O)
{
    __shared__ __align__(32) float sS[4][16 * SROW];  // raw scores, per wave
    __shared__ __align__(32) float sB[4][16];         // per-row broadcast slot

    const int wave  = threadIdx.x >> 5;
    const int lane  = threadIdx.x & 31;
    const int hlf   = lane >> 4;
    const int l15   = lane & 15;
    const int h     = blockIdx.y;
    const int qrow0 = blockIdx.x * 64 + wave * 16;

    const _Float16* Q  = Qh  + h * S_LEN * D_HEAD;
    const _Float16* K  = Kh  + h * S_LEN * D_HEAD;
    const _Float16* VT = VTh + h * D_HEAD * S_LEN;
    float* sw = sS[wave];
    float* sb = sB[wave];

    // Q A-fragments for K-dims 0..31 / 32..63, loaded once
    const _Float16* qp = Q + (qrow0 + l15) * D_HEAD;
    const v16h qa0 = cat8(*(const v8h*)(qp + hlf * 16),      *(const v8h*)(qp + 8 + hlf * 16));
    const v16h qa1 = cat8(*(const v8h*)(qp + 32 + hlf * 16), *(const v8h*)(qp + 40 + hlf * 16));

    // per-lane softmax state for row l15 (duplicated in the lane pair)
    float m_row = -3.0e38f, l_row = 0.0f;
    v8f acc[4] = {};

    for (int j0 = 0; j0 < S_LEN; j0 += 64) {
        // ---- batch-issue ALL demand loads: K fragments first, then V ------
        v16h kb[8], vb[8];
#pragma unroll
        for (int t = 0; t < 4; ++t) {
            const _Float16* kp = K + (j0 + t * 16 + l15) * D_HEAD + hlf * 16;
            kb[2 * t]     = *(const v16h*)kp;          // K-dims  0..31
            kb[2 * t + 1] = *(const v16h*)(kp + 32);   // K-dims 32..63
        }
#pragma unroll
        for (int t = 0; t < 4; ++t) {
            const _Float16* vp = VT + (t * 16 + l15) * S_LEN + j0 + hlf * 16;
            vb[2 * t]     = *(const v16h*)vp;          // keys j0   .. j0+31
            vb[2 * t + 1] = *(const v16h*)(vp + 32);   // keys j0+32.. j0+63
        }

        // ---- prefetch next tile's K/V: one 128B line per lane, 4 instrs ---
        {
            const int jn = (j0 + 64 < S_LEN) ? (j0 + 64) : 0;
            __builtin_prefetch(K + (jn + lane * 2) * D_HEAD, 0, 0);
            __builtin_prefetch(K + (jn + lane * 2 + 1) * D_HEAD, 0, 0);
            __builtin_prefetch(VT + (lane * 2) * S_LEN + jn, 0, 0);
            __builtin_prefetch(VT + (lane * 2 + 1) * S_LEN + jn, 0, 0);
        }

        // ---- scores for 4 key sub-tiles (16 keys each) --------------------
        v8f s[4] = {};
#pragma unroll
        for (int t = 0; t < 4; ++t) {
            s[t] = wmma_f16(qa0, kb[2 * t],     s[t]);
            s[t] = wmma_f16(qa1, kb[2 * t + 1], s[t]);
        }

        // ---- raw scores: C-layout -> row-major LDS ------------------------
        // (same-wave LDS ops are in-order; no barrier needed)
#pragma unroll
        for (int t = 0; t < 4; ++t)
#pragma unroll
            for (int r = 0; r < 8; ++r)
                sw[(r + 8 * hlf) * SROW + t * 16 + l15] = s[t][r];

        // ---- read back this lane's 32 keys of row l15 in A-frag order -----
        float ch[4][8];
        const float* srow = sw + l15 * SROW + hlf * 16;
#pragma unroll
        for (int q = 0; q < 4; ++q) {
            const float4 lo = *(const float4*)(srow + (q & 1) * 8 + (q >> 1) * 32);
            const float4 hi = *(const float4*)(srow + (q & 1) * 8 + (q >> 1) * 32 + 4);
            ch[q][0] = lo.x; ch[q][1] = lo.y; ch[q][2] = lo.z; ch[q][3] = lo.w;
            ch[q][4] = hi.x; ch[q][5] = hi.y; ch[q][6] = hi.z; ch[q][7] = hi.w;
        }

        // ---- softmax (exp2 domain): local tree + one cross-half shuffle ---
        float mx = ch[0][0];
#pragma unroll
        for (int q = 0; q < 4; ++q)
#pragma unroll
            for (int j = 0; j < 8; ++j) mx = fmaxf(mx, ch[q][j]);
        mx = fmaxf(mx, __shfl_xor(mx, 16, 32));
        const float mnew = fmaxf(m_row, mx);
        const float corr = exp2f(m_row - mnew);
        float sum = 0.0f;
#pragma unroll
        for (int q = 0; q < 4; ++q)
#pragma unroll
            for (int j = 0; j < 8; ++j) {
                const float pv = exp2f(ch[q][j] - mnew);
                ch[q][j] = pv;
                sum += pv;
            }
        sum += __shfl_xor(sum, 16, 32);
        l_row = l_row * corr + sum;
        m_row = mnew;

        // ---- rebroadcast corr to C-layout rows via LDS slot ---------------
        if (hlf == 0) sb[l15] = corr;
        const float4 cb0 = *(const float4*)(sb + 8 * hlf);
        const float4 cb1 = *(const float4*)(sb + 8 * hlf + 4);
        const float cc[8] = {cb0.x, cb0.y, cb0.z, cb0.w, cb1.x, cb1.y, cb1.z, cb1.w};
#pragma unroll
        for (int t = 0; t < 4; ++t)
#pragma unroll
            for (int r = 0; r < 8; ++r) acc[t][r] *= cc[r];

        // ---- P already in A-frag order: convert and multiply by V --------
        v16h pa0, pa1;
#pragma unroll
        for (int j = 0; j < 8; ++j) {
            pa0[j]     = (_Float16)ch[0][j];
            pa0[j + 8] = (_Float16)ch[1][j];
            pa1[j]     = (_Float16)ch[2][j];
            pa1[j + 8] = (_Float16)ch[3][j];
        }
#pragma unroll
        for (int t = 0; t < 4; ++t) {
            acc[t] = wmma_f16(pa0, vb[2 * t],     acc[t]);
            acc[t] = wmma_f16(pa1, vb[2 * t + 1], acc[t]);
        }
    }

    // ---- epilogue: broadcast l to C-layout, normalize, store f16 ----------
    if (hlf == 0) sb[l15] = l_row;
    const float4 lb0 = *(const float4*)(sb + 8 * hlf);
    const float4 lb1 = *(const float4*)(sb + 8 * hlf + 4);
    const float lv[8] = {lb0.x, lb0.y, lb0.z, lb0.w, lb1.x, lb1.y, lb1.z, lb1.w};
#pragma unroll
    for (int r = 0; r < 8; ++r) {
        const int   row = qrow0 + r + 8 * hlf;
        const float inv = 1.0f / lv[r];
        _Float16* op = O + row * D_MODEL + h * D_HEAD + l15;
#pragma unroll
        for (int t = 0; t < 4; ++t) op[t * 16] = (_Float16)(acc[t][r] * inv);
    }
}

// ---------------------------------------------------------------------------
// Kernel 3: output projection  out = O @ Wo^T + bo  (f16 x f16 -> fp32 + bias)
// ---------------------------------------------------------------------------
__global__ __launch_bounds__(32)
void out_proj_kernel(const _Float16* __restrict__ O, const _Float16* __restrict__ W,
                     const float* __restrict__ bias, float* __restrict__ out)
{
    const int lane = threadIdx.x & 31;
    const int hlf  = lane >> 4;
    const int l15  = lane & 15;
    const int m0   = blockIdx.x * 32;
    const int nb   = blockIdx.y;

    v8f acc0[4] = {}, acc1[4] = {};
    for (int k0 = 0; k0 < D_MODEL; k0 += 32) {
        const _Float16* ap = O + (m0 + l15) * D_MODEL + k0 + hlf * 16;
        const _Float16* ap1 = ap + 16 * D_MODEL;
        const _Float16* bp = W + (nb * 64 + l15) * D_MODEL + k0 + hlf * 16;
        const v16h a0 = cat8(*(const v8h*)ap,  *(const v8h*)(ap + 8));
        const v16h a1 = cat8(*(const v8h*)ap1, *(const v8h*)(ap1 + 8));
        v16h b[4];
#pragma unroll
        for (int t = 0; t < 4; ++t) b[t] = *(const v16h*)(bp + t * 16 * D_MODEL);
#pragma unroll
        for (int t = 0; t < 4; ++t) {
            acc0[t] = wmma_f16(a0, b[t], acc0[t]);
            acc1[t] = wmma_f16(a1, b[t], acc1[t]);
        }
    }

#pragma unroll
    for (int t = 0; t < 4; ++t) {
        const int   n  = nb * 64 + t * 16 + l15;
        const float bb = bias[n];
#pragma unroll
        for (int r = 0; r < 8; ++r) {
            out[(m0 + r + 8 * hlf) * D_MODEL + n]      = acc0[t][r] + bb;
            out[(m0 + 16 + r + 8 * hlf) * D_MODEL + n] = acc1[t][r] + bb;
        }
    }
}

// ---------------------------------------------------------------------------
extern "C" void kernel_launch(void* const* d_in, const int* in_sizes, int n_in,
                              void* d_out, int out_size, void* d_ws, size_t ws_size,
                              hipStream_t stream)
{
    const float* query = (const float*)d_in[0];
    const float* keyi  = (const float*)d_in[1];
    const float* value = (const float*)d_in[2];
    const float* Wq = (const float*)d_in[3];
    const float* bq = (const float*)d_in[4];
    const float* Wk = (const float*)d_in[5];
    const float* bk = (const float*)d_in[6];
    const float* Wv = (const float*)d_in[7];
    const float* bv = (const float*)d_in[8];
    const float* Wo = (const float*)d_in[9];
    const float* bo = (const float*)d_in[10];
    float* out = (float*)d_out;

    const size_t SD = (size_t)S_LEN * D_MODEL;     // 2M elements
    const size_t DD = (size_t)D_MODEL * D_MODEL;   // 256K elements

    _Float16* p = (_Float16*)d_ws;
    _Float16* xq16 = p; p += SD;      // converted inputs
    _Float16* xk16 = p; p += SD;
    _Float16* xv16 = p; p += SD;
    _Float16* wq16 = p; p += DD;      // converted weights
    _Float16* wk16 = p; p += DD;
    _Float16* wv16 = p; p += DD;
    _Float16* wo16 = p; p += DD;
    _Float16* q16  = p; p += SD;      // projected, per-head
    _Float16* k16  = p; p += SD;
    _Float16* v16  = p; p += SD;      // transposed [H][64][S]
    _Float16* o16  = p; p += SD;      // attention output [S][512]

    // ---- fp32 -> f16 conversion passes ----
    const int gIn = (int)(SD / (8 * 256));   // 1024 blocks
    const int gW  = (int)(DD / (8 * 256));   // 128 blocks
    cvt_f32_f16_kernel<<<gIn, 256, 0, stream>>>(query, xq16);
    cvt_f32_f16_kernel<<<gIn, 256, 0, stream>>>(keyi,  xk16);
    cvt_f32_f16_kernel<<<gIn, 256, 0, stream>>>(value, xv16);
    cvt_f32_f16_kernel<<<gW,  256, 0, stream>>>(Wq, wq16);
    cvt_f32_f16_kernel<<<gW,  256, 0, stream>>>(Wk, wk16);
    cvt_f32_f16_kernel<<<gW,  256, 0, stream>>>(Wv, wv16);
    cvt_f32_f16_kernel<<<gW,  256, 0, stream>>>(Wo, wo16);

    // ---- Q/K/V projections (32x64 strip per wave) ----
    dim3 gproj(S_LEN / 32, D_MODEL / 64);    // 128 x 8
    qkv_proj_kernel<0><<<gproj, 32, 0, stream>>>(xq16, wq16, bq, q16);
    qkv_proj_kernel<1><<<gproj, 32, 0, stream>>>(xk16, wk16, bk, k16);
    qkv_proj_kernel<2><<<gproj, 32, 0, stream>>>(xv16, wv16, bv, v16);

    // ---- flash attention ----
    dim3 gattn(S_LEN / 64, N_HEADS);         // 64 x 8 blocks, 4 waves each
    flash_attn_kernel<<<gattn, 128, 0, stream>>>(q16, k16, v16, o16);

    // ---- output projection ----
    out_proj_kernel<<<gproj, 32, 0, stream>>>(o16, wo16, bo, out);
}
